// TreeTransformer_2405181685799
// MI455X (gfx1250) — matrix operand order repaired and checked by reading, not compiled, not run on hardware
//
#include <hip/hip_runtime.h>

// ---------------- problem constants ----------------
#define NNODES   50000
#define NEDGES   800000
#define DIM      256
#define NETYPES  3
#define NSTEPS   5
#define NGRAPHS  64
#define KTILES   (DIM / 32)      // 8
#define MTILES   (NNODES / 16)   // 3125 (exact)

typedef __bf16 bf16;
typedef bf16  v16bf __attribute__((ext_vector_type(16)));
typedef float v8f   __attribute__((ext_vector_type(8)));

// ---------------- helpers ----------------
static __device__ __forceinline__ unsigned short f32_to_bf16_rne(float f) {
  unsigned u = __float_as_uint(f);
  unsigned r = 0x7FFFu + ((u >> 16) & 1u);
  return (unsigned short)((u + r) >> 16);
}

// A-fragment packed index for a 16x32 bf16 tile (contiguous K order, matching the B packing):
// lanes 0-15: row M=lane, K elems {0..15}; lanes 16-31: row M=lane-16, K {16..31}
static __device__ __forceinline__ size_t apack_idx(int node, int d) {
  int tm = node >> 4, mr = node & 15;
  int tk = d >> 5,  kk = d & 31;
  int lane = mr + ((kk >> 4) << 4);
  int elem = kk & 15;
  return ((((size_t)tm * KTILES + tk) * 32 + lane) << 4) + (size_t)elem;
}

static __device__ __forceinline__ float sigmoidf_(float x) {
  return 1.0f / (1.0f + __expf(-x));
}

static __device__ __forceinline__ void atomicMaxF32(float* addr, float val) {
  unsigned* ua = (unsigned*)addr;
  unsigned old = *ua;
  for (;;) {
    if (__uint_as_float(old) >= val) break;
    unsigned assumed = old;
    old = atomicCAS(ua, assumed, __float_as_uint(val));
    if (old == assumed) break;
  }
}

// ---------------- tiny utility kernels ----------------
__global__ void k_zero(float* p, long n) {
  long i = (long)blockIdx.x * blockDim.x + threadIdx.x;
  long stride = (long)gridDim.x * blockDim.x;
  for (; i < n; i += stride) p[i] = 0.0f;
}

__global__ void k_init_pool(float* gmax, float* denom) {
  int i = threadIdx.x;
  if (i < NGRAPHS) { gmax[i] = -3.402823466e38f; denom[i] = 0.0f; }
}

// ---------------- weight packing (B 32x16 fragment layout) ----------------
// lanes 0-15 hold K=tk*32+elem (col N = tn*16+lane); lanes 16-31 hold K=tk*32+16+elem
__global__ void k_pack_wmsg(const float* __restrict__ W, unsigned short* __restrict__ out) {
  long total = (long)NETYPES * DIM * DIM;
  long i = (long)blockIdx.x * blockDim.x + threadIdx.x;
  long stride = (long)gridDim.x * blockDim.x;
  const int NT = DIM / 16;  // 16
  for (; i < total; i += stride) {
    long e = i / (DIM * DIM);
    long r = i % (DIM * DIM);
    int elem = (int)(r & 15);
    int lane = (int)((r >> 4) & 31);
    long rest = r >> 9;
    int tn = (int)(rest % NT);
    int tk = (int)(rest / NT);
    int k = tk * 32 + (lane & 16) + elem;
    int n = tn * 16 + (lane & 15);
    out[i] = f32_to_bf16_rne(W[e * DIM * DIM + (size_t)k * DIM + n]);
  }
}

// Pack w (J x K row-major, used as B = w^T : [K][J]) into fragment layout.
__global__ void k_pack_wT(const float* __restrict__ w, unsigned short* __restrict__ out,
                          int J, int K) {
  long total = (long)J * K;
  long i = (long)blockIdx.x * blockDim.x + threadIdx.x;
  long stride = (long)gridDim.x * blockDim.x;
  int NT = J / 16;
  for (; i < total; i += stride) {
    int elem = (int)(i & 15);
    int lane = (int)((i >> 4) & 31);
    long rest = i >> 9;
    int tn = (int)(rest % NT);
    int tk = (int)(rest / NT);
    int k = tk * 32 + (lane & 16) + elem;
    int n = tn * 16 + (lane & 15);
    out[i] = f32_to_bf16_rne(w[(size_t)n * K + k]);  // B[k][n] = w[n][k]
  }
}

// ---------------- embedding gather (+ A-fragment pack) ----------------
__global__ void k_embed(const int* __restrict__ x, const float* __restrict__ emb,
                        float* __restrict__ h, unsigned short* __restrict__ hpack) {
  long total = (long)NNODES * DIM;
  long i = (long)blockIdx.x * blockDim.x + threadIdx.x;
  long stride = (long)gridDim.x * blockDim.x;
  for (; i < total; i += stride) {
    int n = (int)(i / DIM), d = (int)(i % DIM);
    float v = emb[(size_t)x[n] * DIM + d];
    h[i] = v;
    hpack[apack_idx(n, d)] = f32_to_bf16_rne(v);
  }
}

__global__ void k_cast_apack(const float* __restrict__ a, unsigned short* __restrict__ apack) {
  long total = (long)NNODES * DIM;
  long i = (long)blockIdx.x * blockDim.x + threadIdx.x;
  long stride = (long)gridDim.x * blockDim.x;
  for (; i < total; i += stride) {
    int n = (int)(i / DIM), d = (int)(i % DIM);
    apack[apack_idx(n, d)] = f32_to_bf16_rne(a[i]);
  }
}

// ---------------- WMMA bf16 GEMM: C = A(MxK) * B(KxN) + bias ----------------
// One wave per 16x64 C strip (1x4 N-blocked). Explicit double-buffering:
// iteration tk+1's fragments are in flight while tk's four WMMAs issue, so
// each WMMA waits only on loads issued a full iteration earlier.
#define WMMA_BF16(A, B, Cacc)                                              \
  __builtin_amdgcn_wmma_f32_16x16x32_bf16(false, (A), false, (B),          \
                                          (short)0, (Cacc), false, false)

__global__ __launch_bounds__(256) void k_wmma_gemm(
    const unsigned short* __restrict__ Apack,   // [Mtiles][8][32][16] bf16
    const unsigned short* __restrict__ Bpack,   // [8][Ntiles][32][16] bf16
    const float* __restrict__ bias,             // [Ncols] (per blockIdx.y slice)
    float* __restrict__ C,                      // [M][Ncols]
    int Mtiles, int NtilesB,                    // NtilesB = Ntiles/4
    int Ntiles, int Ncols,
    long bStrideY, long cStrideY, int biasStrideY) {
  int wave = threadIdx.x >> 5;
  int lane = threadIdx.x & 31;
  int wid = blockIdx.x * 8 + wave;
  if (wid >= Mtiles * NtilesB) return;         // wave-uniform: EXEC stays all-ones
  int tnb = wid % NtilesB;                     // block of 4 N tiles
  int tm  = wid / NtilesB;

  const v16bf* Ap = reinterpret_cast<const v16bf*>(Apack) +
                    (size_t)tm * KTILES * 32 + lane;
  const v16bf* Bp = reinterpret_cast<const v16bf*>(Bpack + (size_t)blockIdx.y * bStrideY) +
                    (size_t)tnb * 4 * 32 + lane;

  v8f acc0 = {}, acc1 = {}, acc2 = {}, acc3 = {};

  // prologue: fetch tk=0 fragments
  v16bf av = Ap[0];
  v16bf b0 = Bp[0];
  v16bf b1 = Bp[32];
  v16bf b2 = Bp[64];
  v16bf b3 = Bp[96];

#pragma unroll
  for (int tk = 0; tk < KTILES; ++tk) {
    v16bf avn = av, b0n = b0, b1n = b1, b2n = b2, b3n = b3;
    if (tk + 1 < KTILES) {                     // prefetch next iteration
      avn = Ap[(size_t)(tk + 1) * 32];
      const v16bf* bpn = Bp + (size_t)(tk + 1) * Ntiles * 32;
      b0n = bpn[0];
      b1n = bpn[32];
      b2n = bpn[64];
      b3n = bpn[96];
    }
    acc0 = WMMA_BF16(av, b0, acc0);
    acc1 = WMMA_BF16(av, b1, acc1);
    acc2 = WMMA_BF16(av, b2, acc2);
    acc3 = WMMA_BF16(av, b3, acc3);
    av = avn; b0 = b0n; b1 = b1n; b2 = b2n; b3 = b3n;
  }

  const float* bi = bias + (size_t)blockIdx.y * biasStrideY;
  float* Cp = C + (size_t)blockIdx.y * cStrideY;
  int mbase = tm * 16 + ((lane >> 4) << 3);    // lanes 16-31 hold rows M=8..15
  int ncol0 = tnb * 64 + (lane & 15);
#pragma unroll
  for (int j = 0; j < 4; ++j) {
    int n = ncol0 + j * 16;
    float bv = bi[n];
    v8f acc = (j == 0) ? acc0 : (j == 1) ? acc1 : (j == 2) ? acc2 : acc3;
#pragma unroll
    for (int i = 0; i < 8; ++i)
      Cp[(size_t)(mbase + i) * Ncols + n] = acc[i] + bv;
  }
}

// ---------------- edge message gather + scatter-add ----------------
// a[dst[e]] += T[etype[e]][src[e]]   (bias already folded into T)
// 64 threads per edge -> edge index is wave-uniform: force scalar loads.
__global__ void k_edge_scatter(const int* __restrict__ src, const int* __restrict__ dst,
                               const int* __restrict__ et, const float* __restrict__ T,
                               float* __restrict__ a) {
  int e = blockIdx.x * 4 + (threadIdx.x >> 6);
  if (e >= NEDGES) return;
  int eu = __builtin_amdgcn_readfirstlane(e);
  int s  = src[eu];
  int dd = dst[eu];
  int t  = et[eu];
  int d = (threadIdx.x & 63) * 4;
  const float4 v = *(const float4*)(T + ((size_t)t * NNODES + s) * DIM + d);
  float* ap = a + (size_t)dd * DIM + d;
  atomicAdd(ap + 0, v.x);
  atomicAdd(ap + 1, v.y);
  atomicAdd(ap + 2, v.z);
  atomicAdd(ap + 3, v.w);
}

// ---------------- GRU elementwise (updates h, re-packs A fragments) ----------------
__global__ void k_gru(const float* __restrict__ gi, const float* __restrict__ gh,
                      float* __restrict__ h, unsigned short* __restrict__ hpack) {
  long total = (long)NNODES * DIM;
  long i = (long)blockIdx.x * blockDim.x + threadIdx.x;
  long stride = (long)gridDim.x * blockDim.x;
  for (; i < total; i += stride) {
    int n = (int)(i / DIM), d = (int)(i % DIM);
    size_t base = (size_t)n * (3 * DIM) + d;
    float ir = gi[base], iz = gi[base + DIM], inn = gi[base + 2 * DIM];
    float hr = gh[base], hz = gh[base + DIM], hn  = gh[base + 2 * DIM];
    float r = sigmoidf_(ir + hr);
    float z = sigmoidf_(iz + hz);
    float nn = tanhf(inn + r * hn);
    float hv = (1.0f - z) * nn + z * h[i];
    h[i] = hv;
    hpack[apack_idx(n, d)] = f32_to_bf16_rne(hv);
  }
}

// ---------------- global attention pooling ----------------
__global__ void k_gate(const float* __restrict__ h, const float* __restrict__ gw,
                       const float* __restrict__ gb, float* __restrict__ gate) {
  int node = (int)((blockIdx.x * (long)blockDim.x + threadIdx.x) >> 5);
  int lane = threadIdx.x & 31;
  if (node >= NNODES) return;
  const float* hp = h + (size_t)node * DIM;
  float s = 0.0f;
#pragma unroll
  for (int d = lane; d < DIM; d += 32) s += hp[d] * gw[d];
  for (int off = 16; off > 0; off >>= 1) s += __shfl_down(s, off, 32);
  if (lane == 0) gate[node] = s + gb[0];
}

__global__ void k_segmax(const float* __restrict__ gate, const int* __restrict__ seg,
                         float* __restrict__ gmax) {
  int n = blockIdx.x * blockDim.x + threadIdx.x;
  if (n < NNODES) atomicMaxF32(&gmax[seg[n]], gate[n]);
}

__global__ void k_expsum(const float* __restrict__ gate, const int* __restrict__ seg,
                         const float* __restrict__ gmax, float* __restrict__ esc,
                         float* __restrict__ denom) {
  int n = blockIdx.x * blockDim.x + threadIdx.x;
  if (n < NNODES) {
    float e = __expf(gate[n] - gmax[seg[n]]);
    esc[n] = e;
    atomicAdd(&denom[seg[n]], e);
  }
}

__global__ void k_outsum(const float* __restrict__ h, const int* __restrict__ seg,
                         const float* __restrict__ esc, const float* __restrict__ denom,
                         float* __restrict__ out) {
  long total = (long)NNODES * DIM;
  long i = (long)blockIdx.x * blockDim.x + threadIdx.x;
  long stride = (long)gridDim.x * blockDim.x;
  for (; i < total; i += stride) {
    int n = (int)(i / DIM), d = (int)(i % DIM);
    int g = seg[n];
    float alpha = esc[n] / denom[g];
    atomicAdd(&out[(size_t)g * DIM + d], h[i] * alpha);
  }
}

// ---------------- host orchestration ----------------
extern "C" void kernel_launch(void* const* d_in, const int* in_sizes, int n_in,
                              void* d_out, int out_size, void* d_ws, size_t ws_size,
                              hipStream_t stream) {
  (void)in_sizes; (void)n_in; (void)out_size; (void)ws_size;
  const int*   x      = (const int*)d_in[0];
  const int*   src    = (const int*)d_in[1];
  const int*   dst    = (const int*)d_in[2];
  const int*   etypes = (const int*)d_in[3];
  const int*   seg    = (const int*)d_in[4];
  const float* emb    = (const float*)d_in[6];
  const float* W_msg  = (const float*)d_in[7];
  const float* b_msg  = (const float*)d_in[8];
  const float* w_ih   = (const float*)d_in[9];
  const float* w_hh   = (const float*)d_in[10];
  const float* b_ih   = (const float*)d_in[11];
  const float* b_hh   = (const float*)d_in[12];
  const float* gate_w = (const float*)d_in[13];
  const float* gate_b = (const float*)d_in[14];
  float* out = (float*)d_out;

  // workspace layout (256B aligned)
  char* ws = (char*)d_ws;
  size_t off = 0;
  auto alloc = [&](size_t bytes) { char* p = ws + off; off += (bytes + 255) & ~(size_t)255; return p; };
  float*          h      = (float*)         alloc((size_t)NNODES * DIM * 4);
  unsigned short* hpack  = (unsigned short*)alloc((size_t)NNODES * DIM * 2);
  float*          a      = (float*)         alloc((size_t)NNODES * DIM * 4);
  unsigned short* apack  = (unsigned short*)alloc((size_t)NNODES * DIM * 2);
  float*          T      = (float*)         alloc((size_t)NETYPES * NNODES * DIM * 4); // aliased as gi
  float*          gi     = T;                                                          // [N][3D]
  float*          gh     = (float*)         alloc((size_t)NNODES * 3 * DIM * 4);
  unsigned short* wmsgP  = (unsigned short*)alloc((size_t)NETYPES * DIM * DIM * 2);
  unsigned short* wihP   = (unsigned short*)alloc((size_t)3 * DIM * DIM * 2);
  unsigned short* whhP   = (unsigned short*)alloc((size_t)3 * DIM * DIM * 2);
  float*          gate   = (float*)         alloc((size_t)NNODES * 4);
  float*          esc    = (float*)         alloc((size_t)NNODES * 4);
  float*          gmax   = (float*)         alloc((size_t)NGRAPHS * 4);
  float*          denom  = (float*)         alloc((size_t)NGRAPHS * 4);

  const dim3 GS(4096), BS(256);

  // pack weights into WMMA B-fragment layouts (every call: deterministic)
  k_pack_wmsg<<<dim3(768), BS, 0, stream>>>(W_msg, wmsgP);
  k_pack_wT<<<dim3(768), BS, 0, stream>>>(w_ih, wihP, 3 * DIM, DIM);
  k_pack_wT<<<dim3(768), BS, 0, stream>>>(w_hh, whhP, 3 * DIM, DIM);

  // h = emb[x]  (f32 + A-fragment bf16)
  k_embed<<<GS, BS, 0, stream>>>(x, emb, h, hpack);

  // wave-strip counts: each wave = 1 Mtile x 4 Ntiles
  const int msgStrips = MTILES * (16 / 4);        // 12500
  const int gruStrips = MTILES * (48 / 4);        // 37500

  for (int step = 0; step < NSTEPS; ++step) {
    // T[e] = h @ W_msg[e] + b_msg[e]   (3x [50000x256]x[256x256], bf16 WMMA)
    k_wmma_gemm<<<dim3((msgStrips + 7) / 8, NETYPES), BS, 0, stream>>>(
        hpack, wmsgP, b_msg, T,
        MTILES, 16 / 4, 16, DIM,
        (long)DIM * DIM, (long)NNODES * DIM, DIM);

    // a = segment_sum over edges of T[etype][src]
    k_zero<<<GS, BS, 0, stream>>>(a, (long)NNODES * DIM);
    k_edge_scatter<<<dim3(NEDGES / 4), BS, 0, stream>>>(src, dst, etypes, T, a);
    k_cast_apack<<<GS, BS, 0, stream>>>(a, apack);

    // gi = a @ w_ih^T + b_ih ; gh = h @ w_hh^T + b_hh   ([50000x256]x[256x768])
    k_wmma_gemm<<<dim3((gruStrips + 7) / 8, 1), BS, 0, stream>>>(
        apack, wihP, b_ih, gi, MTILES, 48 / 4, 48, 3 * DIM, 0, 0, 0);
    k_wmma_gemm<<<dim3((gruStrips + 7) / 8, 1), BS, 0, stream>>>(
        hpack, whhP, b_hh, gh, MTILES, 48 / 4, 48, 3 * DIM, 0, 0, 0);

    // GRU gates -> new h (+ re-pack A fragments for next step)
    k_gru<<<GS, BS, 0, stream>>>(gi, gh, h, hpack);
  }

  // global attention pooling
  k_init_pool<<<dim3(1), dim3(64), 0, stream>>>(gmax, denom);
  k_zero<<<dim3(64), BS, 0, stream>>>(out, (long)NGRAPHS * DIM);
  k_gate<<<dim3((NNODES * 32 + 255) / 256), BS, 0, stream>>>(h, gate_w, gate_b, gate);
  k_segmax<<<dim3((NNODES + 255) / 256), BS, 0, stream>>>(gate, seg, gmax);
  k_expsum<<<dim3((NNODES + 255) / 256), BS, 0, stream>>>(gate, seg, gmax, esc, denom);
  k_outsum<<<GS, BS, 0, stream>>>(h, seg, esc, denom, out);
}
